// FlashMoERouter_51857435132575
// MI455X (gfx1250) — matrix operand order repaired and checked by reading, not compiled, not run on hardware
//
#include <hip/hip_runtime.h>
#include <hip/hip_bf16.h>
#include <math.h>

typedef __bf16 bf16_t;
typedef __attribute__((ext_vector_type(16))) __bf16 v16bf;
typedef __attribute__((ext_vector_type(8)))  float  v8f;

#define DDIM 4096
#define NEXP 64
#define TPB  64            // tokens per block -> 256 workgroups
#define KSTEP 32
#define MT   4             // M tiles (16 rows each) per wave
#define A_STRIDE_BF 40     // bf16 elems per LDS row (80 B: 16B aligned, bank-staggered)
#define A_BUF_ELEMS (TPB * A_STRIDE_BF)   // 2560 bf16 per buffer
#define C_STRIDE 129       // padded f32 stride -> conflict-free column reads

union FragU { v16bf v; int4 q[2]; };

static __device__ __forceinline__ unsigned pack_bf2(float a, float b) {
#if __has_builtin(__builtin_amdgcn_cvt_pk_bf16_f32)
    typedef __attribute__((ext_vector_type(2))) __bf16 v2bf;
    v2bf r = __builtin_amdgcn_cvt_pk_bf16_f32(a, b);   // single v_cvt_pk_bf16_f32
    return __builtin_bit_cast(unsigned, r);
#else
    union { bf16_t h[2]; unsigned u; } t;
    t.h[0] = (bf16_t)a; t.h[1] = (bf16_t)b;
    return t.u;
#endif
}

static __device__ __forceinline__ void sched_fence() {
#if __has_builtin(__builtin_amdgcn_sched_barrier)
    __builtin_amdgcn_sched_barrier(0);   // nothing may be reordered across
#endif
}

// ---- pre-pass: convert gate_w (rows 0..63) and cap_w1 (rows 64..127) to bf16 ----
__global__ void wconvert_kernel(const float* __restrict__ gw,
                                const float* __restrict__ cw,
                                bf16_t* __restrict__ W) {
    int i = (blockIdx.x * blockDim.x + threadIdx.x) * 4;  // over 128*4096 elems
    const float* src = (i < NEXP * DDIM) ? (gw + i) : (cw + (i - NEXP * DDIM));
    float4 f = *(const float4*)src;
    unsigned* dst = (unsigned*)(W + i);
    dst[0] = pack_bf2(f.x, f.y);
    dst[1] = pack_bf2(f.z, f.w);
}

// One pipelined K-step: stage x-tile (fs) to LDS, issue next x loads (fl),
// barrier, prefetch next B (bpref), consume current B (bcons) with 4 WMMAs.
static __device__ __forceinline__
void k_step(bf16_t* __restrict__ buf, const float* __restrict__ xBase, int kk_next,
            const bf16_t* __restrict__ wRow,
            float4 (&fs)[2], float4 (&fl)[2],
            const FragU& bcons, FragU& bpref,
            v8f (&acc)[MT], int tid, int l15, int half) {
    // stage x(it) as bf16 into LDS (consumes fs, loaded one iteration ago)
    #pragma unroll
    for (int p = 0; p < 2; ++p) {
        int lin = p * 1024 + tid * 4;
        int r = lin >> 5, c = lin & 31;
        unsigned* dst = (unsigned*)(buf + r * A_STRIDE_BF + c);
        dst[0] = pack_bf2(fs[p].x, fs[p].y);
        dst[1] = pack_bf2(fs[p].z, fs[p].w);
    }
    // issue x(it+1) loads before the barrier: hidden behind a full iteration
    #pragma unroll
    for (int p = 0; p < 2; ++p) {
        int lin = p * 1024 + tid * 4;
        int r = lin >> 5, c = lin & 31;
        fl[p] = *(const float4*)(xBase + (size_t)r * DDIM + kk_next + c);
    }
    __syncthreads();
    // prefetch B(it+1) into the alternate register set (no loop-carried copy)
    {
        const bf16_t* bp = wRow + kk_next + half * 16;
        bpref.q[0] = *(const int4*)(bp);
        bpref.q[1] = *(const int4*)(bp + 8);
    }
    // load ALL A fragments first, then fence, then the 4 WMMAs:
    // forces 8 outstanding ds_loads -> descending s_wait_dscnt, LDS/WMMA overlap
    FragU af[MT];
    #pragma unroll
    for (int mt = 0; mt < MT; ++mt) {
        const bf16_t* ap = buf + (mt * 16 + l15) * A_STRIDE_BF + half * 16;
        af[mt].q[0] = *(const int4*)(ap);
        af[mt].q[1] = *(const int4*)(ap + 8);
    }
    sched_fence();
    #pragma unroll
    for (int mt = 0; mt < MT; ++mt)
        acc[mt] = __builtin_amdgcn_wmma_f32_16x16x32_bf16(
            false, af[mt].v, false, bcons.v, (short)0, acc[mt], false, false);
}

// ---- fused router kernel ----
__global__ __launch_bounds__(256)
void moe_router_kernel(const float* __restrict__ x,
                       const bf16_t* __restrict__ W,
                       const float* __restrict__ cap_b1,
                       const float* __restrict__ ln_g,
                       const float* __restrict__ ln_b,
                       const float* __restrict__ cap_w2,
                       const float* __restrict__ cap_b2,
                       const float* __restrict__ temperature,
                       const float* __restrict__ usage,
                       float* __restrict__ out) {
    __shared__ __align__(16) float smemC[TPB * C_STRIDE];   // 33024 B
    bf16_t* aBuf = (bf16_t*)smemC;          // double-buffered A staging (2 x 5120 B)

    const int tid  = threadIdx.x;
    const int wave = tid >> 5;
    const int lane = tid & 31;
    const int l15  = lane & 15;
    const int half = lane >> 4;
    const int rowBase = blockIdx.x * TPB;

    // this wave's 16 weight rows (N-stripe); K contiguous per lane
    const bf16_t* wRow  = W + (size_t)(wave * 16 + l15) * DDIM;
    const float*  xBase = x + (size_t)rowBase * DDIM;

    v8f acc[MT];
    #pragma unroll
    for (int i = 0; i < MT; ++i)
        #pragma unroll
        for (int j = 0; j < 8; ++j) acc[i][j] = 0.0f;

    const int NIT = DDIM / KSTEP;   // 128 (even)

    // ---- pipeline prologue: x(0) and B(0) in flight ----
    float4 f0[2], f1[2];
    #pragma unroll
    for (int p = 0; p < 2; ++p) {
        int lin = p * 1024 + tid * 4;           // f32 element index in 64x32 tile
        int r = lin >> 5, c = lin & 31;
        f0[p] = *(const float4*)(xBase + (size_t)r * DDIM + c);
    }
    FragU bA, bB;
    {
        const bf16_t* bp = wRow + half * 16;
        bA.q[0] = *(const int4*)(bp);
        bA.q[1] = *(const int4*)(bp + 8);
    }

    // unroll-by-2 ping-pong: buffers, B regs, and x regs alternate; no copies
    for (int it = 0; it < NIT; it += 2) {
        const int kb   = it * KSTEP;
        const int kkn1 = (kb + KSTEP)     & (DDIM - 1);   // wrapped: in-bounds
        const int kkn2 = (kb + 2 * KSTEP) & (DDIM - 1);
        k_step(aBuf,               xBase, kkn1, wRow, f0, f1, bA, bB,
               acc, tid, l15, half);
        k_step(aBuf + A_BUF_ELEMS, xBase, kkn2, wRow, f1, f0, bB, bA,
               acc, tid, l15, half);
    }

    __syncthreads();   // done with aBuf; smemC becomes the C spill buffer
    #pragma unroll
    for (int mt = 0; mt < MT; ++mt) {
        int rbase = mt * 16 + half * 8;
        int col = wave * 16 + l15;
        #pragma unroll
        for (int r = 0; r < 8; ++r)
            smemC[(rbase + r) * C_STRIDE + col] = acc[mt][r];
    }
    __syncthreads();

    // ---- per-token epilogue: LN -> GELU -> sigmoid capacity -> bias -> top-2 ----
    if (tid < TPB) {
        const float* crow = smemC + tid * C_STRIDE;

        float s = 0.f, s2 = 0.f;
        for (int j = 0; j < NEXP; ++j) {
            float h = crow[64 + j] + cap_b1[j];
            s += h; s2 += h * h;
        }
        float mu  = s * (1.0f / 64.0f);
        float var = s2 * (1.0f / 64.0f) - mu * mu;
        float rinv = rsqrtf(var + 1e-5f);

        float dot = 0.f;
        for (int j = 0; j < NEXP; ++j) {
            float h  = crow[64 + j] + cap_b1[j];
            float hn = (h - mu) * rinv * ln_g[j] + ln_b[j];
            float ge = 0.5f * hn * (1.0f + erff(hn * 0.70710678118f)); // exact GELU
            dot += ge * cap_w2[j];
        }
        float cap = 1.0f / (1.0f + __expf(-(dot + cap_b2[0])));

        float usum = 0.f;
        for (int j = 0; j < NEXP; ++j) usum += usage[j];
        float uinv  = 1.0f / (usum + 1e-6f);
        float scale = cap * fmaxf(temperature[0], 0.1f);

        float v1 = -3.0e38f, v2 = -3.0e38f;
        int   i1 = -1, i2 = -1;
        for (int j = 0; j < NEXP; ++j) {
            float g = (crow[j] - 0.1f * usage[j] * uinv) * scale;
            if (g > v1)      { v2 = v1; i2 = i1; v1 = g; i1 = j; }
            else if (g > v2) { v2 = g;  i2 = j; }
        }
        float rn = 1.0f / (v1 + v2 + 1e-6f);

        float* orow = out + (size_t)(rowBase + tid) * NEXP;
        for (int j4 = 0; j4 < NEXP; j4 += 4) {
            float4 o = make_float4(0.f, 0.f, 0.f, 0.f);
            if (i1 >= j4 && i1 < j4 + 4) ((float*)&o)[i1 - j4] = v1 * rn;
            if (i2 >= j4 && i2 < j4 + 4) ((float*)&o)[i2 - j4] = v2 * rn;
            *(float4*)(orow + j4) = o;
        }
    }
}

extern "C" void kernel_launch(void* const* d_in, const int* in_sizes, int n_in,
                              void* d_out, int out_size, void* d_ws, size_t ws_size,
                              hipStream_t stream) {
    const float* x        = (const float*)d_in[0];
    const float* gate_w   = (const float*)d_in[1];
    const float* cap_w1   = (const float*)d_in[2];
    const float* cap_b1   = (const float*)d_in[3];
    const float* ln_g     = (const float*)d_in[4];
    const float* ln_b     = (const float*)d_in[5];
    const float* cap_w2   = (const float*)d_in[6];
    const float* cap_b2   = (const float*)d_in[7];
    const float* temp     = (const float*)d_in[8];
    const float* usage    = (const float*)d_in[9];

    bf16_t* W = (bf16_t*)d_ws;  // 128*4096 bf16 = 1 MiB

    // 128*4096 elems / 4 per thread / 256 threads = 512 blocks
    wconvert_kernel<<<512, 256, 0, stream>>>(gate_w, cap_w1, W);

    // 16384 tokens / 64 per block = 256 blocks
    moe_router_kernel<<<256, 256, 0, stream>>>(x, W, cap_b1, ln_g, ln_b,
                                               cap_w2, cap_b2, temp, usage,
                                               (float*)d_out);
}
